// HistoryRGCN_39522289058166
// MI455X (gfx1250) — matrix operand order repaired and checked by reading, not compiled, not run on hardware
//
#include <hip/hip_runtime.h>
#include <math.h>

// ---------------- CDNA5 (gfx1250) wave32 WMMA types ----------------
typedef __attribute__((ext_vector_type(16))) __bf16 v16bf;
typedef __attribute__((ext_vector_type(8)))  __bf16 v8bf;
typedef __attribute__((ext_vector_type(8)))  float  v8f;

union ABPack { v16bf v; v8bf h[2]; };

#define NNODES 50000
#define DEGC   16
#define RREL   8
#define FIN    128
#define FHID   128
#define FOUT   40
// Workspace layout (bytes)
#define WB1_OFF   0            // 8*32*32*16 bf16 = 262144 B
#define WB2_OFF   262144       // 3*32*32*16 bf16 =  98304 B
#define BNS_OFF   360448       // 128 f32
#define BNB_OFF   360960       // 128 f32
#define H_OFF     361472       // 50000*128 f32 = 25.6 MB

// ------------------------------------------------------------------
// Prep: swizzle W1/W2 (f32) into the exact bf16 B-fragment layout
// [ct][kt][lane][16] so each lane's B operand is one contiguous 32B
// load.  This code's contiguous k order (same as A, not ISA table order): lane<16 -> N=lane, K=0..15 ;
// lane>=16 -> N=lane-16, K=16..31 (pairs packed per VGPR).
// Also folds BatchNorm into scale/bias.
// ------------------------------------------------------------------
__global__ __launch_bounds__(256)
void prep_kernel(const float* __restrict__ W1, const float* __restrict__ W2,
                 const float* __restrict__ g,  const float* __restrict__ b,
                 const float* __restrict__ mean, const float* __restrict__ var,
                 __bf16* __restrict__ wb1, __bf16* __restrict__ wb2,
                 float* __restrict__ bnScale, float* __restrict__ bnBias)
{
    int gid = blockIdx.x * 256 + threadIdx.x;
    if (gid < 131072) {                       // WB1: ct(8) kt(32) lane(32) j(16)
        int j    = gid & 15;
        int lane = (gid >> 4) & 31;
        int kt   = (gid >> 9) & 31;
        int ct   =  gid >> 14;
        int K = kt * 32 + (lane & 16) + j;    // (lane&16)==16 iff lane>=16
        int n = ct * 16 + (lane & 15);
        wb1[gid] = (__bf16)W1[K * FHID + n];  // W1 flat: [r*128+d][128] == [K][n]
    } else if (gid < 131072 + 49152) {        // WB2: ct(3) kt(32) lane(32) j(16)
        int f = gid - 131072;
        int j    = f & 15;
        int lane = (f >> 4) & 31;
        int kt   = (f >> 9) & 31;
        int ct   =  f >> 14;
        int K = kt * 32 + (lane & 16) + j;
        int n = ct * 16 + (lane & 15);
        wb2[f] = (n < FOUT) ? (__bf16)W2[K * FOUT + n] : (__bf16)0.0f;  // pad 40->48
    } else if (gid < 131072 + 49152 + 128) {  // BN fold
        int c = gid - 131072 - 49152;
        float sc = g[c] * rsqrtf(var[c] + 1e-5f);
        bnScale[c] = sc;
        bnBias[c]  = b[c] - mean[c] * sc;
    }
}

// ------------------------------------------------------------------
// One RGCN layer.  blockIdx.x = tile of 16 destination nodes.
// 256 threads = 8 wave32s.
//   Phase A1: gather-accumulate neighbor rows into relation-bucketed
//             S[16][R*128] f32 in LDS (aggregate-then-transform).
//   Phase A2: S * (1/deg) -> bf16 A tile in WMMA A layout order.
//   Phase B : per-wave 16x16 output tile, K=1024 via 32 bf16 WMMAs.
// ------------------------------------------------------------------
template <bool LAYER2>
__global__ __launch_bounds__(256)
void rgcn_layer_kernel(const float* __restrict__ feat,   // [N,128] f32 (x or h)
                       const __bf16* __restrict__ wb,    // [CT][32][32][16] bf16
                       const int* __restrict__ ptr,
                       const int* __restrict__ idx,
                       const int* __restrict__ ety,
                       const float* __restrict__ bnScale,
                       const float* __restrict__ bnBias,
                       float* __restrict__ out)          // L1: h[N,128]; L2: [N,40]
{
    __shared__ __align__(16) float  S[16][RREL * FIN];     // 64 KB f32 accumulators
    __shared__ __align__(32) __bf16 Abf[16][RREL * FIN];   // 32 KB bf16 A tile
    __shared__ float invdeg[16];
    __shared__ float bnS[FHID], bnB[FHID];
    __shared__ float outT[16][48];

    const int tid  = threadIdx.x;
    const int lane = tid & 31;
    const int w    = tid >> 5;          // wave id 0..7
    const int tile = blockIdx.x;        // 16 nodes per tile (50000/16 = 3125)

    for (int t = tid; t < 16 * RREL * FIN; t += 256) ((float*)S)[t] = 0.0f;
    if (!LAYER2 && tid < FHID) { bnS[tid] = bnScale[tid]; bnB[tid] = bnBias[tid]; }
    __syncthreads();

    // ---- Phase A1: wave w handles local nodes 2w and 2w+1 ----
    const float4* f4 = (const float4*)feat;           // 128 f32 = 32 float4 / row
    for (int s = 0; s < 2; ++s) {
        int ln   = 2 * w + s;
        int node = tile * 16 + ln;
        int p0 = ptr[node], p1 = ptr[node + 1];
        int deg = p1 - p0;
        if (lane == 0) invdeg[ln] = 1.0f / (float)max(deg, 1);
        int eI = 0, eT = 0;
        if (lane < deg) { eI = idx[p0 + lane]; eT = ety[p0 + lane]; }
        for (int e = 0; e < deg; ++e) {
            int src = __shfl(eI, e, 32);
            int r   = __shfl(eT, e, 32);
            float4 v = f4[(size_t)src * 32 + lane];   // L2-resident gather, b128
            float* dst = &S[ln][r * FIN + lane * 4];
            atomicAdd(dst + 0, v.x);                  // ds_add_f32, wave-ordered
            atomicAdd(dst + 1, v.y);
            atomicAdd(dst + 2, v.z);
            atomicAdd(dst + 3, v.w);
        }
    }
    __syncthreads();

    // ---- Phase A2: fold 1/deg, convert to bf16 ----
    for (int t = tid; t < 16 * RREL * FIN; t += 256) {
        int row = t >> 10;
        ((__bf16*)Abf)[t] = (__bf16)(((const float*)S)[t] * invdeg[row]);
    }
    __syncthreads();

    // ---- Phase B: WMMA. wave w -> output column tile ct = w ----
    const int NCT = LAYER2 ? 3 : 8;     // 48 (padded) vs 128 output cols
    const int hiA = lane >> 4;          // 0: lanes 0-15, 1: lanes 16-31
    if (w < NCT) {
        const int ct = w;
        v8f acc = {};
        // A layout (16-bit 16x32): lane<16: K {0..15}; lane>=16: {16..31} (matches B packing)
        const __bf16* abase = &Abf[lane & 15][0];
        const __bf16* bbase = wb + (((size_t)ct * 32 * 32) + lane) * 16;
        for (int kt = 0; kt < 32; ++kt) {
            ABPack a, b;
            a.h[0] = *(const v8bf*)(abase + kt * 32 + hiA * 16);       // ds_load_b128
            a.h[1] = *(const v8bf*)(abase + kt * 32 + hiA * 16 + 8);   // ds_load_b128
            b.v    = *(const v16bf*)(bbase + (size_t)kt * 32 * 16);    // 32B global
            acc = __builtin_amdgcn_wmma_f32_16x16x32_bf16(
                      false, a.v, false, b.v, (short)0, acc, false, false);
        }
        // C/D layout: lane<16 -> rows 0..7 = acc[0..7], N=lane ; lane>=16 -> rows 8..15
        const int col   = ct * 16 + (lane & 15);
        const int mbase = hiA * 8;
        if (!LAYER2) {
            float sc = bnS[col], bi = bnB[col];
#pragma unroll
            for (int v = 0; v < 8; ++v) {
                float val = fmaxf(acc[v] * sc + bi, 0.0f);   // BN + ReLU
                out[(size_t)(tile * 16 + mbase + v) * FHID + col] = val;
            }
        } else {
#pragma unroll
            for (int v = 0; v < 8; ++v) outT[mbase + v][col] = acc[v];
        }
    }

    if (LAYER2) {
        __syncthreads();
        if (tid < 16) {                       // log_softmax over 40 logits / row
            float mx = -INFINITY;
            for (int j = 0; j < FOUT; ++j) mx = fmaxf(mx, outT[tid][j]);
            float sum = 0.0f;
            for (int j = 0; j < FOUT; ++j) sum += __expf(outT[tid][j] - mx);
            float lse = mx + __logf(sum);
            size_t base = (size_t)(tile * 16 + tid) * FOUT;
            for (int j = 0; j < FOUT; ++j) out[base + j] = outT[tid][j] - lse;
        }
    }
}

// ------------------------------------------------------------------
extern "C" void kernel_launch(void* const* d_in, const int* in_sizes, int n_in,
                              void* d_out, int out_size, void* d_ws, size_t ws_size,
                              hipStream_t stream)
{
    const float* x   = (const float*)d_in[0];
    const float* W1  = (const float*)d_in[1];
    const float* W2  = (const float*)d_in[2];
    const float* bng = (const float*)d_in[3];
    const float* bnb = (const float*)d_in[4];
    const float* bnm = (const float*)d_in[5];
    const float* bnv = (const float*)d_in[6];
    const int*   ptr = (const int*)d_in[7];
    const int*   idx = (const int*)d_in[8];
    const int*   ety = (const int*)d_in[9];
    float* out = (float*)d_out;

    char* ws = (char*)d_ws;
    __bf16* wb1    = (__bf16*)(ws + WB1_OFF);
    __bf16* wb2    = (__bf16*)(ws + WB2_OFF);
    float* bnScale = (float*)(ws + BNS_OFF);
    float* bnBias  = (float*)(ws + BNB_OFF);
    float* h       = (float*)(ws + H_OFF);

    // 131072 + 49152 + 128 = 180352 work items
    prep_kernel<<<705, 256, 0, stream>>>(W1, W2, bng, bnb, bnm, bnv,
                                         wb1, wb2, bnScale, bnBias);
    rgcn_layer_kernel<false><<<NNODES / 16, 256, 0, stream>>>(
        x, wb1, ptr, idx, ety, bnScale, bnBias, h);
    rgcn_layer_kernel<true><<<NNODES / 16, 256, 0, stream>>>(
        h, wb2, ptr, idx, ety, nullptr, nullptr, out);
}
